// Block_49744311222996
// MI455X (gfx1250) — compile-verified, matrix operand order checked by reading, not run on hardware
//
#include <hip/hip_runtime.h>

// ---------------------------------------------------------------------------
// Types for CDNA5 WMMA (gfx1250, wave32)
// ---------------------------------------------------------------------------
typedef __attribute__((ext_vector_type(16))) __bf16 v16bf;
typedef __attribute__((ext_vector_type(8)))  float  v8f;
typedef int v4i __attribute__((vector_size(16)));   // matches async-LDS builtin param

#define AS1 __attribute__((address_space(1)))
#define AS3 __attribute__((address_space(3)))

#if defined(__AMDGCN__) && __has_builtin(__builtin_amdgcn_global_load_async_to_lds_b128) && \
    __has_builtin(__builtin_amdgcn_s_wait_asynccnt)
#define USE_ASYNC_LDS 1
#else
#define USE_ASYNC_LDS 0
#endif

#define Bsz 4
#define Tsz 2048
#define Csz 1024
#define Hn  16
#define HSz 64
#define CUT 1024   // CTX/2^LAYER = 1024, min(T, t)

__device__ __forceinline__ unsigned short f2bf(float x) {
    union { float f; unsigned int u; } c; c.f = x;
    unsigned int r = c.u + 0x7FFFu + ((c.u >> 16) & 1u);
    return (unsigned short)(r >> 16);
}

__device__ __forceinline__ float gelu_exact(float x) {
    return 0.5f * x * (1.0f + erff(x * 0.70710678118654752f));
}

__device__ __forceinline__ v8f wmma_bf16(v16bf a, v16bf b, v8f c) {
    // v_wmma_f32_16x16x32_bf16: D = A(16x32) * B(32x16) + C(16x16 f32)
    return __builtin_amdgcn_wmma_f32_16x16x32_bf16(
        false, a, false, b, (short)0, c, false, false);
}

// Load a 16x32 bf16 fragment. Works for A (lane=row m, contraction=k) and for
// B when the tile is stored transposed [n][k] (lane=col n, contraction=k).
// 16-bit A-matrix layout (ISA 7.12.2): lanes 0-15 -> K {0..7,16..23},
// lanes 16-31 -> K {8..15,24..31}; pairs of K per VGPR.
__device__ __forceinline__ v16bf load_frag(const unsigned short* __restrict__ base,
                                           int stride, int kOff) {
    const int l   = threadIdx.x & 31;
    const int idx = l & 15;
    const int kh  = l >> 4;
    union { v16bf v; unsigned int u[8]; } f;
#pragma unroll
    for (int v = 0; v < 8; ++v) {
        int kk = ((v & 4) << 2) + ((v & 3) << 1) + (kh << 3);
        f.u[v] = *(const unsigned int*)(base + (size_t)idx * stride + kOff + kk);
    }
    return f.v;
}

// ---------------------------------------------------------------------------
// LayerNorm (fp32 in) -> bf16 out.  One block per row of 1024.
// ---------------------------------------------------------------------------
__global__ __launch_bounds__(256) void ln_kernel(
    const float* __restrict__ X, const float* __restrict__ w,
    const float* __restrict__ bb, unsigned short* __restrict__ out) {
    const long long row = blockIdx.x;
    const float* xr = X + row * (long long)Csz;
    const int tid = threadIdx.x, lane = tid & 31, wave = tid >> 5;
    float vals[4], s = 0.f, sq = 0.f;
#pragma unroll
    for (int i = 0; i < 4; ++i) {
        float v = xr[tid + i * 256];
        vals[i] = v; s += v; sq += v * v;
    }
#pragma unroll
    for (int off = 16; off > 0; off >>= 1) {
        s  += __shfl_xor(s,  off, 32);
        sq += __shfl_xor(sq, off, 32);
    }
    __shared__ float sm[18];
    if (lane == 0) { sm[wave] = s; sm[8 + wave] = sq; }
    __syncthreads();
    if (tid == 0) {
        float ts = 0.f, tq = 0.f;
#pragma unroll
        for (int i = 0; i < 8; ++i) { ts += sm[i]; tq += sm[8 + i]; }
        float mean = ts * (1.0f / Csz);
        float var  = tq * (1.0f / Csz) - mean * mean;
        sm[16] = mean;
        sm[17] = rsqrtf(var + 1e-5f);
    }
    __syncthreads();
    const float mean = sm[16], rstd = sm[17];
#pragma unroll
    for (int i = 0; i < 4; ++i) {
        int c = tid + i * 256;
        out[row * Csz + c] = f2bf((vals[i] - mean) * rstd * w[c] + bb[c]);
    }
}

// ---------------------------------------------------------------------------
// fp32 -> bf16 conversions
// ---------------------------------------------------------------------------
__global__ void cvt_mat(const float* __restrict__ in,
                        unsigned short* __restrict__ out, int n) {
    int i = blockIdx.x * blockDim.x + threadIdx.x;
    if (i < n) out[i] = f2bf(in[i]);
}

// Repack (H,C,HS) -> [C, H*HS] so head-concat == column order
__global__ void cvt_qkv(const float* __restrict__ in,
                        unsigned short* __restrict__ out) {
    int o = blockIdx.x * blockDim.x + threadIdx.x;  // o = c*1024 + h*64 + d
    int c = o >> 10, rem = o & 1023, h = rem >> 6, d = rem & 63;
    out[o] = f2bf(in[(size_t)h * Csz * HSz + (size_t)c * HSz + d]);
}

// ---------------------------------------------------------------------------
// Tiled bf16 GEMM:  Out[M,N] = A[M,K] @ Bw[K,N] + bias (+ residual)
//   block tile 128x64, BK=32, 8 waves each owning 32x32 (2x2 WMMA tiles)
//   A-tile staged with GLOBAL_LOAD_ASYNC_TO_LDS_B128 (ASYNCcnt) when available
// EPI: 0 = bf16 out + bias, 1 = bf16 out + bias + exact GELU,
//      2 = f32 out + bias + residual (residual has its own row mapping)
// ---------------------------------------------------------------------------
template <int EPI>
__global__ __launch_bounds__(256) void gemm_bf16(
    const unsigned short* __restrict__ A, int lda, int RB, long long SB, long long aBase,
    const unsigned short* __restrict__ Bw, int ldb,
    const float* __restrict__ bias, void* __restrict__ Out,
    const float* __restrict__ Res, int RBr, long long SBr, long long rBase,
    int M, int N, int K) {
    __shared__ unsigned short As[128 * 32];   // 8 KB
    __shared__ unsigned short BsT[64 * 32];   // 4 KB (transposed [n][k])

    const int tid = threadIdx.x, lane = tid & 31, wave = tid >> 5;
    const int wm = wave & 3, wn = wave >> 2;
    const int lcol = lane & 15, kh = lane >> 4;
    const int bN = blockIdx.x, bM = blockIdx.y;

    v8f acc[2][2];
#pragma unroll
    for (int i = 0; i < 2; ++i)
#pragma unroll
        for (int j = 0; j < 2; ++j)
#pragma unroll
            for (int r = 0; r < 8; ++r) acc[i][j][r] = 0.f;

    for (int k0 = 0; k0 < K; k0 += 32) {
        // --- stage A tile 128x32 (async direct-to-LDS when available) ---
#pragma unroll
        for (int i = 0; i < 2; ++i) {
            int c = tid + i * 256;
            int r = c >> 2, col = (c & 3) << 3;
            int Rg = bM * 128 + r;
            long long off = aBase + (long long)(Rg / RB) * SB +
                            (long long)(Rg % RB) * lda + k0 + col;
#if USE_ASYNC_LDS
            __builtin_amdgcn_global_load_async_to_lds_b128(
                (AS1 v4i*)(A + off), (AS3 v4i*)(&As[r * 32 + col]), 0, 0);
#else
            *(uint4*)(&As[r * 32 + col]) = *(const uint4*)(A + off);
#endif
        }
        // --- stage B tile 32x64, transposed into BsT[n][k] ---
        {
            int r = tid >> 3, col = (tid & 7) << 3;
            uint4 d = *(const uint4*)(Bw + (long long)(k0 + r) * ldb + bN * 64 + col);
            const unsigned short* s = (const unsigned short*)&d;
#pragma unroll
            for (int j = 0; j < 8; ++j) BsT[(col + j) * 32 + r] = s[j];
            // prefetch next iteration's B chunk into cache
            if (k0 + 32 < K)
                __builtin_prefetch(Bw + (long long)(k0 + 32 + r) * ldb + bN * 64 + col, 0, 3);
        }
#if USE_ASYNC_LDS
        __builtin_amdgcn_s_wait_asynccnt(0);
#endif
        __syncthreads();

        v16bf af[2];
#pragma unroll
        for (int mi = 0; mi < 2; ++mi)
            af[mi] = load_frag(&As[(wm * 32 + mi * 16) * 32], 32, 0);
#pragma unroll
        for (int ni = 0; ni < 2; ++ni) {
            v16bf bf_ = load_frag(&BsT[(wn * 32 + ni * 16) * 32], 32, 0);
#pragma unroll
            for (int mi = 0; mi < 2; ++mi)
                acc[mi][ni] = wmma_bf16(af[mi], bf_, acc[mi][ni]);
        }
        __syncthreads();
    }

    // --- epilogue ---
#pragma unroll
    for (int mi = 0; mi < 2; ++mi)
#pragma unroll
        for (int ni = 0; ni < 2; ++ni)
#pragma unroll
            for (int r = 0; r < 8; ++r) {
                int m   = wm * 32 + mi * 16 + r + 8 * kh;
                int Rg  = bM * 128 + m;
                int col = bN * 64 + wn * 32 + ni * 16 + lcol;
                float v = acc[mi][ni][r] + bias[col];
                if (EPI == 0) {
                    ((unsigned short*)Out)[(long long)Rg * N + col] = f2bf(v);
                } else if (EPI == 1) {
                    ((unsigned short*)Out)[(long long)Rg * N + col] = f2bf(gelu_exact(v));
                } else {
                    long long roff = rBase + (long long)(Rg / RBr) * SBr +
                                     (long long)(Rg % RBr) * N;
                    ((float*)Out)[(long long)Rg * N + col] = v + Res[roff + col];
                }
            }
}

// ---------------------------------------------------------------------------
// Flash attention: grid (CUT/128, H, B), 256 threads = 8 waves.
// Each wave: 16 query rows.  K streamed via GLOBAL_LOAD_ASYNC_TO_LDS_B128,
// V staged transposed; online softmax in fp32; P round-trips LDS as bf16.
// scale = C^-0.5 = 1/32 (reference scales by n_embd**-0.5).
// ---------------------------------------------------------------------------
__global__ __launch_bounds__(256) void attn_kernel(
    const unsigned short* __restrict__ Q, const unsigned short* __restrict__ Kb,
    const unsigned short* __restrict__ Vb, unsigned short* __restrict__ O) {
    __shared__ unsigned short Ks[64 * 64];    // [key][d]   8 KB
    __shared__ unsigned short VsT[64 * 64];   // [d][key]   8 KB
    __shared__ unsigned short Ps[8 * 16 * 64];// per-wave P 16 KB

    const int tid = threadIdx.x, lane = tid & 31, wave = tid >> 5;
    const int lcol = lane & 15, kh = lane >> 4;
    const int bq = blockIdx.x, h = blockIdx.y, b = blockIdx.z;

    const int qRow0 = bq * 128 + wave * 16;                 // in [0, CUT)
    const long long qGlobRow = (long long)b * CUT + qRow0;  // Q/O row
    const int qposMax = (Tsz - CUT) + qRow0 + 15;           // wave's max key

    // Q fragments (16 rows x 64 HS), loaded once, directly from global
    const unsigned short* qbase = Q + qGlobRow * Csz + h * HSz;
    v16bf qfrag[2];
#pragma unroll
    for (int kc = 0; kc < 2; ++kc) qfrag[kc] = load_frag(qbase, Csz, kc * 32);

    float mrun[8], lrun[8];
    v8f oacc[4];
#pragma unroll
    for (int r = 0; r < 8; ++r) { mrun[r] = -INFINITY; lrun[r] = 0.f; }
#pragma unroll
    for (int nt = 0; nt < 4; ++nt)
#pragma unroll
        for (int r = 0; r < 8; ++r) oacc[nt][r] = 0.f;

    const int qposMaxBlock = (Tsz - CUT) + bq * 128 + 127;
    int nTiles = qposMaxBlock / 64 + 1;
    if (nTiles > Tsz / 64) nTiles = Tsz / 64;

    for (int kt = 0; kt < nTiles; ++kt) {
        const int keyBase = kt * 64;
        // cooperative stage of K (async row-major) and V (transposed)
#pragma unroll
        for (int i = 0; i < 2; ++i) {
            int c = tid + i * 256;
            int r = c >> 3, col = (c & 7) << 3;
            long long gr = ((long long)b * Tsz + keyBase + r) * Csz + h * HSz + col;
#if USE_ASYNC_LDS
            __builtin_amdgcn_global_load_async_to_lds_b128(
                (AS1 v4i*)(Kb + gr), (AS3 v4i*)(&Ks[r * 64 + col]), 0, 0);
#else
            *(uint4*)(&Ks[r * 64 + col]) = *(const uint4*)(Kb + gr);
#endif
            uint4 vd = *(const uint4*)(Vb + gr);
            const unsigned short* vs = (const unsigned short*)&vd;
#pragma unroll
            for (int j = 0; j < 8; ++j) VsT[(col + j) * 64 + r] = vs[j];
            // prefetch next key tile's V rows
            if (keyBase + 64 < Tsz)
                __builtin_prefetch(Vb + gr + 64 * Csz, 0, 3);
        }
#if USE_ASYNC_LDS
        __builtin_amdgcn_s_wait_asynccnt(0);
#endif
        __syncthreads();

        if (keyBase <= qposMax) {
            // S = Q K^T  (16 x 64 keys)
            v8f s[4];
#pragma unroll
            for (int nt = 0; nt < 4; ++nt) {
#pragma unroll
                for (int r = 0; r < 8; ++r) s[nt][r] = 0.f;
#pragma unroll
                for (int kc = 0; kc < 2; ++kc) {
                    v16bf kf = load_frag(&Ks[(nt * 16) * 64], 64, kc * 32);
                    s[nt] = wmma_bf16(qfrag[kc], kf, s[nt]);
                }
            }
            // scale + causal mask + row max
            float rmax[8];
#pragma unroll
            for (int r = 0; r < 8; ++r) rmax[r] = -INFINITY;
#pragma unroll
            for (int nt = 0; nt < 4; ++nt)
#pragma unroll
                for (int r = 0; r < 8; ++r) {
                    int key = keyBase + nt * 16 + lcol;
                    int qp  = (Tsz - CUT) + qRow0 + r + 8 * kh;
                    float v = s[nt][r] * 0.03125f;           // C^-0.5
                    v = (key <= qp) ? v : -INFINITY;
                    s[nt][r] = v;
                    rmax[r] = fmaxf(rmax[r], v);
                }
#pragma unroll
            for (int r = 0; r < 8; ++r)
#pragma unroll
                for (int off = 1; off < 16; off <<= 1)
                    rmax[r] = fmaxf(rmax[r], __shfl_xor(rmax[r], off, 32));
            // online softmax update
            float scl[8], rsum[8];
#pragma unroll
            for (int r = 0; r < 8; ++r) {
                float mnew = fmaxf(mrun[r], rmax[r]);
                scl[r]  = __expf(mrun[r] - mnew);
                mrun[r] = mnew;
                lrun[r] *= scl[r];
                rsum[r] = 0.f;
            }
            unsigned short* Pw = &Ps[wave * 1024];
#pragma unroll
            for (int nt = 0; nt < 4; ++nt)
#pragma unroll
                for (int r = 0; r < 8; ++r) {
                    float p = __expf(s[nt][r] - mrun[r]);
                    rsum[r] += p;
                    Pw[(r + 8 * kh) * 64 + nt * 16 + lcol] = f2bf(p);
                }
#pragma unroll
            for (int r = 0; r < 8; ++r) {
#pragma unroll
                for (int off = 1; off < 16; off <<= 1)
                    rsum[r] += __shfl_xor(rsum[r], off, 32);
                lrun[r] += rsum[r];
            }
            // rescale O and accumulate P V
#pragma unroll
            for (int nt = 0; nt < 4; ++nt)
#pragma unroll
                for (int r = 0; r < 8; ++r) oacc[nt][r] *= scl[r];
            v16bf pa[2];
#pragma unroll
            for (int kc = 0; kc < 2; ++kc) pa[kc] = load_frag(Pw, 64, kc * 32);
#pragma unroll
            for (int nt = 0; nt < 4; ++nt)
#pragma unroll
                for (int kc = 0; kc < 2; ++kc) {
                    v16bf vf = load_frag(&VsT[(nt * 16) * 64], 64, kc * 32);
                    oacc[nt] = wmma_bf16(pa[kc], vf, oacc[nt]);
                }
        }
        __syncthreads();
    }

    // normalize + store bf16 (col order h*64+d == head concat)
#pragma unroll
    for (int nt = 0; nt < 4; ++nt)
#pragma unroll
        for (int r = 0; r < 8; ++r) {
            int m = r + 8 * kh;
            float v = oacc[nt][r] / lrun[r];
            O[(qGlobRow + m) * Csz + h * HSz + nt * 16 + lcol] = f2bf(v);
        }
}

// ---------------------------------------------------------------------------
// Host launcher
// ---------------------------------------------------------------------------
extern "C" void kernel_launch(void* const* d_in, const int* in_sizes, int n_in,
                              void* d_out, int out_size, void* d_ws, size_t ws_size,
                              hipStream_t stream) {
    (void)in_sizes; (void)n_in; (void)out_size; (void)ws_size;
    const float* x      = (const float*)d_in[0];
    const float* ln1_w  = (const float*)d_in[1];
    const float* ln1_b  = (const float*)d_in[2];
    const float* Wq     = (const float*)d_in[3];
    const float* bq     = (const float*)d_in[4];
    const float* Wk     = (const float*)d_in[5];
    const float* bk     = (const float*)d_in[6];
    const float* Wv     = (const float*)d_in[7];
    const float* bv     = (const float*)d_in[8];
    const float* proj_w = (const float*)d_in[9];
    const float* proj_b = (const float*)d_in[10];
    const float* ln2_w  = (const float*)d_in[11];
    const float* ln2_b  = (const float*)d_in[12];
    const float* lin1_w = (const float*)d_in[13];
    const float* lin1_b = (const float*)d_in[14];
    const float* lin2_w = (const float*)d_in[15];
    const float* lin2_b = (const float*)d_in[16];
    float* out = (float*)d_out;

    char* p = (char*)d_ws;
    auto carve = [&](size_t bytes) -> void* {
        void* r = (void*)p;
        p += (bytes + 255) & ~(size_t)255;
        return r;
    };
    const size_t BT  = (size_t)Bsz * Tsz;      // 8192
    const size_t BC  = (size_t)Bsz * CUT;      // 4096
    unsigned short* h_bf   = (unsigned short*)carve(BT * Csz * 2);
    unsigned short* wq2    = (unsigned short*)carve((size_t)Csz * Csz * 2);
    unsigned short* wk2    = (unsigned short*)carve((size_t)Csz * Csz * 2);
    unsigned short* wv2    = (unsigned short*)carve((size_t)Csz * Csz * 2);
    unsigned short* pjw    = (unsigned short*)carve((size_t)Csz * Csz * 2);
    unsigned short* l1w    = (unsigned short*)carve((size_t)Csz * 4 * Csz * 2);
    unsigned short* l2w    = (unsigned short*)carve((size_t)4 * Csz * Csz * 2);
    unsigned short* qbuf   = (unsigned short*)carve(BC * Csz * 2);
    unsigned short* kbuf   = (unsigned short*)carve(BT * Csz * 2);
    unsigned short* vbuf   = (unsigned short*)carve(BT * Csz * 2);
    unsigned short* obuf   = (unsigned short*)carve(BC * Csz * 2);
    float*          resb   = (float*)carve(BC * Csz * 4);
    unsigned short* h2_bf  = (unsigned short*)carve(BC * Csz * 2);
    unsigned short* ff1_bf = (unsigned short*)carve(BC * 4 * Csz * 2);

    // 1) LN1 over [B*T, C] -> bf16
    ln_kernel<<<dim3(BT), 256, 0, stream>>>(x, ln1_w, ln1_b, h_bf);

    // 2) weight conversion / repack
    const int M1 = Csz * Csz;
    cvt_qkv<<<M1 / 256, 256, 0, stream>>>(Wq, wq2);
    cvt_qkv<<<M1 / 256, 256, 0, stream>>>(Wk, wk2);
    cvt_qkv<<<M1 / 256, 256, 0, stream>>>(Wv, wv2);
    cvt_mat<<<M1 / 256, 256, 0, stream>>>(proj_w, pjw, M1);
    cvt_mat<<<4 * M1 / 256, 256, 0, stream>>>(lin1_w, l1w, 4 * M1);
    cvt_mat<<<4 * M1 / 256, 256, 0, stream>>>(lin2_w, l2w, 4 * M1);

    const long long strideBT = (long long)Tsz * Csz;
    const long long cropBase = (long long)(Tsz - CUT) * Csz;

    // 3) K, V over all T rows; Q over cropped rows
    gemm_bf16<0><<<dim3(16, 64), 256, 0, stream>>>(
        h_bf, Csz, (int)BT, 0, 0, wk2, Csz, bk, kbuf,
        nullptr, 1, 0, 0, (int)BT, Csz, Csz);
    gemm_bf16<0><<<dim3(16, 64), 256, 0, stream>>>(
        h_bf, Csz, (int)BT, 0, 0, wv2, Csz, bv, vbuf,
        nullptr, 1, 0, 0, (int)BT, Csz, Csz);
    gemm_bf16<0><<<dim3(16, 32), 256, 0, stream>>>(
        h_bf, Csz, CUT, strideBT, cropBase, wq2, Csz, bq, qbuf,
        nullptr, 1, 0, 0, (int)BC, Csz, Csz);

    // 4) flash attention
    attn_kernel<<<dim3(CUT / 128, Hn, Bsz), 256, 0, stream>>>(qbuf, kbuf, vbuf, obuf);

    // 5) proj + residual-from-x -> res (fp32)
    gemm_bf16<2><<<dim3(16, 32), 256, 0, stream>>>(
        obuf, Csz, (int)BC, 0, 0, pjw, Csz, proj_b, resb,
        x, CUT, strideBT, cropBase, (int)BC, Csz, Csz);

    // 6) LN2 -> bf16
    ln_kernel<<<dim3(BC), 256, 0, stream>>>(resb, ln2_w, ln2_b, h2_bf);

    // 7) lin1 + exact GELU -> bf16
    gemm_bf16<1><<<dim3(64, 32), 256, 0, stream>>>(
        h2_bf, Csz, (int)BC, 0, 0, l1w, 4 * Csz, lin1_b, ff1_bf,
        nullptr, 1, 0, 0, (int)BC, 4 * Csz, Csz);

    // 8) lin2 + residual-from-res -> d_out (fp32)
    gemm_bf16<2><<<dim3(16, 32), 256, 0, stream>>>(
        ff1_bf, 4 * Csz, (int)BC, 0, 0, l2w, Csz, lin2_b, out,
        resb, (int)BC, 0, 0, (int)BC, Csz, 4 * Csz);
}
